// RAR_60344290508844
// MI455X (gfx1250) — matrix operand order repaired and checked by reading, not compiled, not run on hardware
//
#include <hip/hip_runtime.h>

// ---------------------------------------------------------------------------
// Problem dims (from the reference)
// ---------------------------------------------------------------------------
#define T_STEPS 64
#define BATCH   512
#define OBS_DIM 256
#define ACT_DIM 32
#define HID     1024
#define FEAT    512
#define GRU_IN  (OBS_DIM + ACT_DIM)   // 288
#define GATES   (3 * HID)             // 3072
#define LOG2PI  1.8378770664093453f

typedef unsigned short ushort_t;
typedef __attribute__((ext_vector_type(16))) __bf16 v16bf;
typedef __attribute__((ext_vector_type(8)))  float  v8f;

__device__ __forceinline__ ushort_t f2bf(float f) {
    unsigned u = __float_as_uint(f);
    unsigned r = u + 0x7fffu + ((u >> 16) & 1u);   // round-to-nearest-even
    return (ushort_t)(r >> 16);
}

// ---------------------------------------------------------------------------
// fp32 -> bf16 weight conversion (weights stay N x K row-major == torch layout)
// ---------------------------------------------------------------------------
__global__ void k_f32_to_bf16(const float* __restrict__ s, ushort_t* __restrict__ d, int n) {
    int i = blockIdx.x * blockDim.x + threadIdx.x;
    if (i < n) d[i] = f2bf(s[i]);
}

// ---------------------------------------------------------------------------
// WMMA GEMM core: C[M,N] = A[M,K] * W[N,K]^T (+bias)(+ELU)(+fused loss)
//  - bf16 inputs, f32 WMMA accumulate (v_wmma_f32_16x16x32_bf16)
//  - 128x128 tile / 256-thread block (8 waves), K stepped by 32
//  - double-buffered LDS (one workgroup barrier per K-slab)
//  - FLAGS is a compile-time constant -> straight-line epilogue
// ---------------------------------------------------------------------------
enum { GF_BIAS = 1, GF_ELU = 2, GF_F32 = 4, GF_BF16 = 8, GF_LOSS = 16 };

struct GemmProb {
    const ushort_t* A;      // M x K bf16 row-major
    const ushort_t* W;      // N x K bf16 row-major
    const float*    bias;   // N
    float*          Cf;     // M x N f32 out
    ushort_t*       Cb;     // M x N bf16 out
    const float*    tgt;    // loss target, same layout as Cf
    float*          lacc;   // loss accumulator
    int             N, K;
};

template<int FLAGS>
__device__ __forceinline__ void gemm_body(const GemmProb p,
                                          ushort_t (* __restrict__ As)[128 * 40],
                                          ushort_t (* __restrict__ Ws)[128 * 40])
{
    const int tid  = threadIdx.x;
    const int lane = tid & 31;
    const int wv   = tid >> 5;
    const int wm   = wv & 3;            // 32-row block within tile
    const int wn   = wv >> 2;           // 64-col block within tile
    const int m0   = blockIdx.y * 128;
    const int n0   = blockIdx.x * 128;
    const int ldr  = tid >> 1;          // 0..127 : tile row to stage
    const int ldh  = tid & 1;           // 16-element half of the 32-wide K slab
    const int N    = p.N;
    const int K    = p.K;

    v8f acc[2][4];
    const v8f vzero = {0.f, 0.f, 0.f, 0.f, 0.f, 0.f, 0.f, 0.f};
#pragma unroll
    for (int i = 0; i < 2; ++i)
#pragma unroll
        for (int j = 0; j < 4; ++j) acc[i][j] = vzero;

    union FragU { uint4 q[2]; v16bf v; };

    const uint4* gA = reinterpret_cast<const uint4*>(p.A + (size_t)(m0 + ldr) * K + ldh * 16);
    const uint4* gW = reinterpret_cast<const uint4*>(p.W + (size_t)(n0 + ldr) * K + ldh * 16);

    // prologue: stage slab 0 into buffer 0
    {
        uint4 a0 = gA[0], a1 = gA[1];
        uint4 w0 = gW[0], w1 = gW[1];
        uint4* sa = reinterpret_cast<uint4*>(&As[0][ldr * 40 + ldh * 16]);
        sa[0] = a0; sa[1] = a1;
        uint4* sw = reinterpret_cast<uint4*>(&Ws[0][ldr * 40 + ldh * 16]);
        sw[0] = w0; sw[1] = w1;
    }
    __syncthreads();

    const int hl  = lane >> 4;          // lane half
    const int l16 = lane & 15;

    int cur = 0;
    for (int k0 = 0; k0 < K; k0 += 32) {
        const bool more = (k0 + 32) < K;
        uint4 na0, na1, nw0, nw1;
        if (more) {
            // next slab: global_load_b128 x4 overlapping with the WMMAs below
            const uint4* ga = gA + (k0 + 32) / 8;   // 8 ushorts per uint4
            na0 = ga[0]; na1 = ga[1];
            const uint4* gw = gW + (k0 + 32) / 8;
            nw0 = gw[0]; nw1 = gw[1];
            if (k0 + 64 < K) {                      // CDNA5 prefetch path
                __builtin_prefetch(p.A + (size_t)(m0 + ldr) * K + k0 + 64 + ldh * 16, 0, 1);
                __builtin_prefetch(p.W + (size_t)(n0 + ldr) * K + k0 + 64 + ldh * 16, 0, 1);
            }
        }

        // A fragments: lane (m = l16), K runs [8*hl, 8*hl+8) and +16
        FragU afr[2];
#pragma unroll
        for (int i = 0; i < 2; ++i) {
            const int m = wm * 32 + i * 16 + l16;
            afr[i].q[0] = *reinterpret_cast<const uint4*>(&As[cur][m * 40 + hl * 8]);
            afr[i].q[1] = *reinterpret_cast<const uint4*>(&As[cur][m * 40 + 16 + hl * 8]);
        }
        // B fragments: lane (n = l16), K runs [8*hl, 8*hl+8) and +16
        FragU bfr[4];
#pragma unroll
        for (int j = 0; j < 4; ++j) {
            const int n = wn * 64 + j * 16 + l16;
            const uint4* q = reinterpret_cast<const uint4*>(&Ws[cur][n * 40 + hl * 8]);
            bfr[j].q[0] = q[0];
            bfr[j].q[1] = q[2];
        }
        // 8 WMMAs per wave per K-slab
#pragma unroll
        for (int i = 0; i < 2; ++i)
#pragma unroll
            for (int j = 0; j < 4; ++j)
                acc[i][j] = __builtin_amdgcn_wmma_f32_16x16x32_bf16(
                    false, afr[i].v, false, bfr[j].v, (short)0, acc[i][j], false, false);

        if (more) {                                  // fill the other buffer
            uint4* sa = reinterpret_cast<uint4*>(&As[cur ^ 1][ldr * 40 + ldh * 16]);
            sa[0] = na0; sa[1] = na1;
            uint4* sw = reinterpret_cast<uint4*>(&Ws[cur ^ 1][ldr * 40 + ldh * 16]);
            sw[0] = nw0; sw[1] = nw1;
        }
        __syncthreads();                             // single barrier per slab
        cur ^= 1;
    }

    // -------- epilogue (fully specialized, straight-line)
    float lsum = 0.0f;
#pragma unroll
    for (int i = 0; i < 2; ++i) {
#pragma unroll
        for (int j = 0; j < 4; ++j) {
            const int ncol = n0 + wn * 64 + j * 16 + l16;
            const float bv = (FLAGS & GF_BIAS) ? p.bias[ncol] : 0.0f;
#pragma unroll
            for (int r = 0; r < 8; ++r) {
                const int mrow = m0 + wm * 32 + i * 16 + hl * 8 + r;
                float v = acc[i][j][r] + bv;
                if (FLAGS & GF_ELU) v = (v > 0.0f) ? v : (__expf(v) - 1.0f);
                const size_t idx = (size_t)mrow * N + ncol;
                if (FLAGS & GF_F32)  p.Cf[idx] = v;
                if (FLAGS & GF_BF16) p.Cb[idx] = f2bf(v);
                if (FLAGS & GF_LOSS) {
                    const float d = p.tgt[idx] - v;
                    lsum += 0.5f * (d * d + LOG2PI);
                }
            }
        }
    }
    if (FLAGS & GF_LOSS) {
#pragma unroll
        for (int m = 16; m > 0; m >>= 1) lsum += __shfl_xor(lsum, m, 32);
        if (lane == 0) atomicAdd(p.lacc, lsum);
    }
}

// Dual-problem launcher: blockIdx.z picks a fully-specialized body; the two
// inlined bodies share one static LDS allocation.
template<int F0, int F1>
__global__ __launch_bounds__(256)
void k_gemm_dual(GemmProb p0, GemmProb p1)
{
    __shared__ __align__(16) ushort_t As[2][128 * 40];
    __shared__ __align__(16) ushort_t Ws[2][128 * 40];
    if (blockIdx.z == 0) gemm_body<F0>(p0, As, Ws);
    else                 gemm_body<F1>(p1, As, Ws);
}

template<int F0>
__global__ __launch_bounds__(256)
void k_gemm_single(GemmProb p0)
{
    __shared__ __align__(16) ushort_t As[2][128 * 40];
    __shared__ __align__(16) ushort_t Ws[2][128 * 40];
    gemm_body<F0>(p0, As, Ws);
}

// ---------------------------------------------------------------------------
// GRU gate nonlinearity: h' = (1-z)*n + z*h  (torch GRUCell semantics)
// ---------------------------------------------------------------------------
__global__ void k_gru_gates(const float* __restrict__ gi, const float* __restrict__ gh,
                            const float* __restrict__ bih, const float* __restrict__ bhh,
                            float* __restrict__ h, ushort_t* __restrict__ hbf)
{
    const int idx = blockIdx.x * blockDim.x + threadIdx.x;   // BATCH*HID exact
    const int b = idx >> 10;
    const int j = idx & 1023;
    const size_t base = (size_t)b * GATES;
    const float ir = gi[base + j]           + bih[j];
    const float iz = gi[base + HID + j]     + bih[HID + j];
    const float in = gi[base + 2 * HID + j] + bih[2 * HID + j];
    const float hr = gh[base + j]           + bhh[j];
    const float hz = gh[base + HID + j]     + bhh[HID + j];
    const float hn = gh[base + 2 * HID + j] + bhh[2 * HID + j];
    const float r = 1.0f / (1.0f + __expf(-(ir + hr)));
    const float z = 1.0f / (1.0f + __expf(-(iz + hz)));
    const float n = tanhf(in + r * hn);
    const float hv = (1.0f - z) * n + z * h[idx];
    h[idx]   = hv;
    hbf[idx] = f2bf(hv);
}

// ---------------------------------------------------------------------------
// Reward head: N=1 dot product per batch row (one wave32 per row) + loss
// ---------------------------------------------------------------------------
__global__ void k_rew_head(const float* __restrict__ a1, const float* __restrict__ w2,
                           const float* __restrict__ b2, const float* __restrict__ rew_t,
                           float* __restrict__ out_t, float* __restrict__ acc)
{
    const int lane = threadIdx.x & 31;
    const int wv   = threadIdx.x >> 5;
    const int row  = blockIdx.x * 8 + wv;
    const float* ap = a1 + (size_t)row * FEAT;
    float s = 0.0f;
#pragma unroll
    for (int k = 0; k < 16; ++k) s += ap[lane * 16 + k] * w2[lane * 16 + k];
#pragma unroll
    for (int m = 16; m > 0; m >>= 1) s += __shfl_xor(s, m, 32);
    if (lane == 0) {
        const float mean = s + b2[0];
        out_t[row] = mean;
        const float d = rew_t[row] - mean;
        atomicAdd(acc, 0.5f * (d * d + LOG2PI));
    }
}

// ---------------------------------------------------------------------------
// Build bf16 GRU input x = [obs_mean_t, action_t] (and the t=0 variant)
// ---------------------------------------------------------------------------
__global__ void k_pack_x(const float* __restrict__ mean_t, const float* __restrict__ act_t,
                         ushort_t* __restrict__ x)
{
    const int idx = blockIdx.x * blockDim.x + threadIdx.x;
    if (idx >= BATCH * GRU_IN) return;
    const int b = idx / GRU_IN, c = idx % GRU_IN;
    const float v = (c < OBS_DIM) ? mean_t[b * OBS_DIM + c]
                                  : act_t[b * ACT_DIM + (c - OBS_DIM)];
    x[idx] = f2bf(v);
}

__global__ void k_pack_x0(const float* __restrict__ obs0, ushort_t* __restrict__ x)
{
    const int idx = blockIdx.x * blockDim.x + threadIdx.x;
    if (idx >= BATCH * GRU_IN) return;
    const int b = idx / GRU_IN, c = idx % GRU_IN;
    x[idx] = (c < OBS_DIM) ? f2bf(obs0[b * OBS_DIM + c]) : (ushort_t)0;
}

__global__ void k_finalize(const float* __restrict__ acc, float* __restrict__ out)
{
    out[0] = (acc[0] + acc[1]) / (float)(T_STEPS * BATCH);
}

// ---------------------------------------------------------------------------
// Host-side orchestration (all on `stream`, graph-capture safe)
// ---------------------------------------------------------------------------
extern "C" void kernel_launch(void* const* d_in, const int* in_sizes, int n_in,
                              void* d_out, int out_size, void* d_ws, size_t ws_size,
                              hipStream_t stream)
{
    (void)in_sizes; (void)n_in; (void)out_size; (void)ws_size;

    const float* obs    = (const float*)d_in[0];
    const float* action = (const float*)d_in[1];
    const float* reward = (const float*)d_in[2];
    const float* w_ih   = (const float*)d_in[3];
    const float* w_hh   = (const float*)d_in[4];
    const float* b_ih   = (const float*)d_in[5];
    const float* b_hh   = (const float*)d_in[6];
    const float* ow0 = (const float*)d_in[7];  const float* ob0 = (const float*)d_in[8];
    const float* ow1 = (const float*)d_in[9];  const float* ob1 = (const float*)d_in[10];
    const float* ow2 = (const float*)d_in[11]; const float* ob2 = (const float*)d_in[12];
    const float* rw0 = (const float*)d_in[13]; const float* rb0 = (const float*)d_in[14];
    const float* rw1 = (const float*)d_in[15]; const float* rb1 = (const float*)d_in[16];
    const float* rw2 = (const float*)d_in[17]; const float* rb2 = (const float*)d_in[18];

    float* out      = (float*)d_out;
    float* loss_out = out;
    float* pre_obs  = out + 1;
    float* pre_rew  = out + 1 + (size_t)T_STEPS * BATCH * OBS_DIM;

    // workspace carve-up (256B aligned)
    char* ws = (char*)d_ws;
    size_t off = 0;
    auto alloc = [&](size_t bytes) -> void* {
        void* p = ws + off;
        off += (bytes + 255) & ~(size_t)255;
        return p;
    };
    ushort_t* wih_bf = (ushort_t*)alloc((size_t)GATES * GRU_IN * 2);
    ushort_t* whh_bf = (ushort_t*)alloc((size_t)GATES * HID * 2);
    ushort_t* ow0_bf = (ushort_t*)alloc((size_t)FEAT * HID * 2);
    ushort_t* ow1_bf = (ushort_t*)alloc((size_t)FEAT * FEAT * 2);
    ushort_t* ow2_bf = (ushort_t*)alloc((size_t)OBS_DIM * FEAT * 2);
    ushort_t* rw0_bf = (ushort_t*)alloc((size_t)FEAT * HID * 2);
    ushort_t* rw1_bf = (ushort_t*)alloc((size_t)FEAT * FEAT * 2);
    float*    h_f    = (float*)   alloc((size_t)BATCH * HID * 4);
    ushort_t* h_bf   = (ushort_t*)alloc((size_t)BATCH * HID * 2);
    ushort_t* x_bf   = (ushort_t*)alloc((size_t)BATCH * GRU_IN * 2);
    float*    gi     = (float*)   alloc((size_t)BATCH * GATES * 4);
    float*    gh     = (float*)   alloc((size_t)BATCH * GATES * 4);
    ushort_t* a0o_bf = (ushort_t*)alloc((size_t)BATCH * FEAT * 2);
    ushort_t* a1o_bf = (ushort_t*)alloc((size_t)BATCH * FEAT * 2);
    ushort_t* a0r_bf = (ushort_t*)alloc((size_t)BATCH * FEAT * 2);
    float*    a1r_f  = (float*)   alloc((size_t)BATCH * FEAT * 4);
    float*    accs   = (float*)   alloc(2 * sizeof(float));

    hipMemsetAsync(accs, 0, 2 * sizeof(float), stream);
    hipMemsetAsync(h_f,  0, (size_t)BATCH * HID * 4, stream);
    hipMemsetAsync(h_bf, 0, (size_t)BATCH * HID * 2, stream);

    auto conv = [&](const float* s, ushort_t* d, int n) {
        k_f32_to_bf16<<<(n + 255) / 256, 256, 0, stream>>>(s, d, n);
    };
    conv(w_ih, wih_bf, GATES * GRU_IN);
    conv(w_hh, whh_bf, GATES * HID);
    conv(ow0,  ow0_bf, FEAT * HID);
    conv(ow1,  ow1_bf, FEAT * FEAT);
    conv(ow2,  ow2_bf, OBS_DIM * FEAT);
    conv(rw0,  rw0_bf, FEAT * HID);
    conv(rw1,  rw1_bf, FEAT * FEAT);

    const dim3 blk(256);
    const int MB = BATCH / 128;   // 4 m-blocks

    auto gru_step = [&]() {
        // gi = x @ Wih^T  ||  gh = h @ Whh^T  (one launch, z selects problem)
        GemmProb pi = { x_bf, wih_bf, nullptr, gi, nullptr, nullptr, nullptr,
                        GATES, GRU_IN };
        GemmProb ph = { h_bf, whh_bf, nullptr, gh, nullptr, nullptr, nullptr,
                        GATES, HID };
        k_gemm_dual<GF_F32, GF_F32>
            <<<dim3(GATES / 128, MB, 2), blk, 0, stream>>>(pi, ph);
        k_gru_gates<<<(BATCH * HID) / 256, 256, 0, stream>>>(gi, gh, b_ih, b_hh, h_f, h_bf);
    };

    // h0 = GRU([obs[0], 0], h=0)
    k_pack_x0<<<(BATCH * GRU_IN + 255) / 256, 256, 0, stream>>>(obs, x_bf);
    gru_step();

    for (int t = 0; t < T_STEPS; ++t) {
        float*       mean_t = pre_obs + (size_t)t * BATCH * OBS_DIM;
        const float* obs_t  = obs     + (size_t)t * BATCH * OBS_DIM;
        const float* act_t  = action  + (size_t)t * BATCH * ACT_DIM;
        const float* rew_t  = reward  + (size_t)t * BATCH;
        float*       prew_t = pre_rew + (size_t)t * BATCH;

        // layer0 (obs || rew): elu(h @ W0^T + b0)
        {
            GemmProb po = { h_bf, ow0_bf, ob0, nullptr, a0o_bf, nullptr, nullptr,
                            FEAT, HID };
            GemmProb pr = { h_bf, rw0_bf, rb0, nullptr, a0r_bf, nullptr, nullptr,
                            FEAT, HID };
            k_gemm_dual<GF_BIAS | GF_ELU | GF_BF16, GF_BIAS | GF_ELU | GF_BF16>
                <<<dim3(FEAT / 128, MB, 2), blk, 0, stream>>>(po, pr);
        }
        // layer1 (obs || rew): elu(a0 @ W1^T + b1)
        {
            GemmProb po = { a0o_bf, ow1_bf, ob1, nullptr, a1o_bf, nullptr, nullptr,
                            FEAT, FEAT };
            GemmProb pr = { a0r_bf, rw1_bf, rb1, a1r_f, nullptr, nullptr, nullptr,
                            FEAT, FEAT };
            k_gemm_dual<GF_BIAS | GF_ELU | GF_BF16, GF_BIAS | GF_ELU | GF_F32>
                <<<dim3(FEAT / 128, MB, 2), blk, 0, stream>>>(po, pr);
        }
        // obs head: mean -> d_out, fused Gaussian log-prob reduction
        {
            GemmProb po = { a1o_bf, ow2_bf, ob2, mean_t, nullptr, obs_t, &accs[0],
                            OBS_DIM, FEAT };
            k_gemm_single<GF_BIAS | GF_F32 | GF_LOSS>
                <<<dim3(OBS_DIM / 128, MB, 1), blk, 0, stream>>>(po);
        }
        // reward head (N=1 -> wave dot kernel, fused loss)
        k_rew_head<<<BATCH / 8, 256, 0, stream>>>(a1r_f, rw2, rb2, rew_t, prew_t, &accs[1]);

        // next hidden state: x = [obs_mean_t, action_t]
        k_pack_x<<<(BATCH * GRU_IN + 255) / 256, 256, 0, stream>>>(mean_t, act_t, x_bf);
        gru_step();
    }

    k_finalize<<<1, 1, 0, stream>>>(accs, loss_out);
}
